// IntraRNN_41798621725284
// MI455X (gfx1250) — matrix operand order repaired and checked by reading, not compiled, not run on hardware
//
#include <hip/hip_runtime.h>

// ---------------------------------------------------------------------------
// MI455X (gfx1250) implementation of the IntraRNN attention+GRU+projection
// pipeline. All GEMMs use V_WMMA_F32_16X16X32_F16 (wave32 WMMA).
// ---------------------------------------------------------------------------

typedef __attribute__((ext_vector_type(16))) _Float16 v16h;
typedef __attribute__((ext_vector_type(8)))  _Float16 v8h;
typedef __attribute__((ext_vector_type(8)))  float    v8f;

#define Bdim 2048
#define Mdim 64
#define Hdim 256
#define Gdim 768           // 3*H gates
#define Vdim 50000
#define LDA  264           // padded LDS row stride in halfs (16B-aligned rows)

__device__ __forceinline__ v8f wmma_16x16x32(v16h a, v16h b, v8f c) {
  // 8 args: (neg_a, A, neg_b, B, c_mod, C, reuse_a, reuse_b)
  return __builtin_amdgcn_wmma_f32_16x16x32_f16(false, a, false, b, (short)0, c,
                                                false, false);
}

// A fragment (16x32, f16) from LDS, row-major [m][k], stride lda halfs.
// Contiguous k order used here: lane L holds row M=L%16; element e -> K = e + 16*(L/16).
__device__ __forceinline__ v16h load_a_frag(const _Float16* As, int lda,
                                            int m0, int k0, int lane) {
  const int ph = lane >> 4, r = lane & 15;
  const _Float16* p = As + (size_t)(m0 + r) * lda + k0 + 16 * ph;
  v8h lo = *(const v8h*)p;          // K = k0 + 16*ph + 0..7  (ds_read_b128)
  v8h hi = *(const v8h*)(p + 8);    // K = k0 + 16*ph + 8..15 (ds_read_b128)
  v16h a;
#pragma unroll
  for (int i = 0; i < 8; ++i) { a[i] = lo[i]; a[i + 8] = hi[i]; }
  return a;
}

// B fragment (32x16) from global f32 B[k][n], row-major ldb (n contiguous).
// Contiguous k order used here: lane L holds col N=L%16; element e -> K = e + 16*(L/16).
// Addresses for a fixed e: lanes 0-15 read 16 consecutive floats of one row,
// lanes 16-31 of another row -> 2 cachelines per load, fully coalesced.
__device__ __forceinline__ v16h load_b_frag(const float* Bp, int ldb,
                                            int k0, int n0, int lane) {
  const int ph = lane >> 4, c = lane & 15;
  const float* p = Bp + (size_t)(k0 + 16 * ph) * ldb + n0 + c;
  v16h b;
#pragma unroll
  for (int e = 0; e < 16; ++e) b[e] = (_Float16)p[(size_t)e * ldb];
  return b;
}

// B fragment for D = X @ W^T with W row-major [n][k] (k contiguous, ldw).
// Each lane reads 16 contiguous floats of row W[n0 + L%16].
__device__ __forceinline__ v16h load_bT_frag(const float* Wp, int ldw,
                                             int k0, int n0, int lane, int nmax) {
  const int ph = lane >> 4, c = lane & 15;
  int n = n0 + c; if (n > nmax) n = nmax;   // clamp for ragged N edge
  const float* p = Wp + (size_t)n * ldw + k0 + 16 * ph;
  v16h b;
#pragma unroll
  for (int e = 0; e < 16; ++e) b[e] = (_Float16)p[e];
  return b;
}

// ---------------------------------------------------------------------------
// Kernel 1: per-batch attention. One block (8 waves) per batch element.
// ---------------------------------------------------------------------------
__global__ __launch_bounds__(256) void attn_kernel(
    const float* __restrict__ inter_output,   // B*M*H
    const float* __restrict__ hidden,         // B*H (hidden[0])
    const float* __restrict__ W_inter, const float* __restrict__ b_inter,
    const float* __restrict__ W_hidden, const float* __restrict__ b_hidden,
    const float* __restrict__ W_scale, const float* __restrict__ b_scale,
    const float* __restrict__ W_lt1, const float* __restrict__ b_lt1,
    const int*  __restrict__ user_list,
    float* __restrict__ attn_out,             // B*M
    float* __restrict__ context)              // B*H (workspace)
{
  __shared__ __attribute__((aligned(16))) _Float16 Ms[Mdim * LDA]; // A, then io
  __shared__ float h0s[Hdim], v0s[Hdim], v1s[Hdim], wsc[Hdim];
  __shared__ float energ[Mdim], attns[Mdim];

  const int b = blockIdx.x, tid = threadIdx.x;
  const int lane = tid & 31, w = tid >> 5;
  const int u = user_list[b];
  const float* A   = inter_output + (size_t)b * Mdim * Hdim;
  const float* Wi  = W_inter  + (size_t)u * Hdim * Hdim;
  const float* Wh  = W_hidden + (size_t)u * Hdim * Hdim;
  const float* Wl  = W_lt1    + (size_t)u * 2 * Hdim * Hdim;
  const float* bi  = b_inter  + (size_t)u * Hdim;
  const float* bhv = b_hidden + (size_t)u * Hdim;
  const float* blv = b_lt1    + (size_t)u * Hdim;
  const float* Wsv = W_scale  + (size_t)u * Hdim;
  const float  bsc = b_scale[u];

  // Stage inter_output -> f16 LDS; h0, W_scale -> LDS; zero energies.
  for (int i = tid; i < Mdim * Hdim; i += 256) {
    int m = i >> 8, k = i & 255;
    Ms[m * LDA + k] = (_Float16)A[i];
  }
  h0s[tid] = hidden[(size_t)b * Hdim + tid];
  wsc[tid] = Wsv[tid];
  if (tid < Mdim) energ[tid] = 0.f;
  __syncthreads();

  // GEMV v0 = h0 @ W_hidden + b_hidden   (thread tid = output column)
  {
    float acc = bhv[tid];
    for (int h = 0; h < Hdim; ++h) acc += h0s[h] * Wh[(size_t)h * Hdim + tid];
    v0s[tid] = acc;
  }
  __syncthreads();
  // GEMV v1 = v0 @ W_lt1[0:H] + b_lt1  (the broadcast ht_a path of `cat`)
  {
    float acc = blv[tid];
    for (int h = 0; h < Hdim; ++h) acc += v0s[h] * Wl[(size_t)h * Hdim + tid];
    v1s[tid] = acc;
  }

  // Wave tiling: mt = M-tile (4), nh = N-half; each wave: 1 M-tile x 8 N-tiles.
  const int mt = w & 3, nh = w >> 2, m0 = mt * 16;
  const int cc = lane & 15, ph = lane >> 4;
  const v8f vzero = {0.f, 0.f, 0.f, 0.f, 0.f, 0.f, 0.f, 0.f};
  v8f acc[8];
#pragma unroll
  for (int t = 0; t < 8; ++t) acc[t] = vzero;

  // GEMM1: io = A @ W_inter   (K = 256, 8 K-steps of 32)
  for (int ks = 0; ks < 8; ++ks) {
    const int k0 = ks * 32;
    v16h af = load_a_frag(Ms, LDA, m0, k0, lane);
#pragma unroll
    for (int t = 0; t < 8; ++t) {
      v16h bf = load_b_frag(Wi, Hdim, k0, (nh * 8 + t) * 16, lane);
      acc[t] = wmma_16x16x32(af, bf, acc[t]);
    }
  }
  __syncthreads();                 // all waves done reading A from Ms
  // Store io (+b_inter) back into Ms as f16 (D layout: M=m0+j+8*ph, N=cc)
#pragma unroll
  for (int t = 0; t < 8; ++t) {
    const int n = (nh * 8 + t) * 16 + cc;
    const float bv = bi[n];
#pragma unroll
    for (int j = 0; j < 8; ++j)
      Ms[(size_t)(m0 + j + 8 * ph) * LDA + n] = (_Float16)(acc[t][j] + bv);
  }
  __syncthreads();

  // GEMM2: pre = io @ W_lt1[H:2H]; result = tanh(pre + v1); energy partials.
#pragma unroll
  for (int t = 0; t < 8; ++t) acc[t] = vzero;
  const float* Wlb = Wl + (size_t)Hdim * Hdim;
  for (int ks = 0; ks < 8; ++ks) {
    const int k0 = ks * 32;
    v16h af = load_a_frag(Ms, LDA, m0, k0, lane);
#pragma unroll
    for (int t = 0; t < 8; ++t) {
      v16h bf = load_b_frag(Wlb, Hdim, k0, (nh * 8 + t) * 16, lane);
      acc[t] = wmma_16x16x32(af, bf, acc[t]);
    }
  }
  float ep[8];
#pragma unroll
  for (int j = 0; j < 8; ++j) ep[j] = 0.f;
#pragma unroll
  for (int t = 0; t < 8; ++t) {
    const int n = (nh * 8 + t) * 16 + cc;
    const float vv = v1s[n], wn = wsc[n];
#pragma unroll
    for (int j = 0; j < 8; ++j) ep[j] += tanhf(acc[t][j] + vv) * wn;
  }
#pragma unroll
  for (int j = 0; j < 8; ++j) {      // reduce over the 16 N-lanes of each half
    float p = ep[j];
    p += __shfl_xor(p, 8, 16);
    p += __shfl_xor(p, 4, 16);
    p += __shfl_xor(p, 2, 16);
    p += __shfl_xor(p, 1, 16);
    if (cc == 0) atomicAdd(&energ[m0 + j + 8 * ph], p);
  }
  __syncthreads();

  // Softmax over M=64 (redundant per-thread scan of 64 LDS floats is cheap).
  float mx = -3.0e38f;
  for (int m = 0; m < Mdim; ++m) mx = fmaxf(mx, energ[m] + bsc);
  float sm = 0.f;
  for (int m = 0; m < Mdim; ++m) sm += expf(energ[m] + bsc - mx);
  if (tid < Mdim) {
    const float aw = expf(energ[tid] + bsc - mx) / sm;
    attns[tid] = aw;
    attn_out[(size_t)b * Mdim + tid] = aw;
  }
  __syncthreads();

  // context[n] = sum_m attn[m] * inter_output[b][m][n]  (full fp32 from global)
  {
    float c = 0.f;
    for (int m = 0; m < Mdim; ++m) c += attns[m] * A[(size_t)m * Hdim + tid];
    context[(size_t)b * Hdim + tid] = c;
  }
}

// ---------------------------------------------------------------------------
// Kernel 2: GRU gate GEMMs. GI = [emb|context] @ W_ih^T ; GH = h0 @ W_hh^T.
// Grid: (B/64, 768/128). Wave: 1 M-tile x 4 N-tiles.
// ---------------------------------------------------------------------------
__global__ __launch_bounds__(256) void gru_gemm_kernel(
    const float* __restrict__ emb, const float* __restrict__ context,
    const float* __restrict__ hidden,
    const float* __restrict__ W_ih,   // 768 x 512, row-major
    const float* __restrict__ W_hh,   // 768 x 256, row-major
    float* __restrict__ gi, float* __restrict__ gh)
{
  __shared__ __attribute__((aligned(16))) _Float16 Xs[Mdim * LDA];
  const int bb = blockIdx.x * Mdim;
  const int nbase = blockIdx.y * 128;
  const int tid = threadIdx.x, lane = tid & 31, w = tid >> 5;
  const int mt = w & 3, nh = w >> 2, m0 = mt * 16;
  const int cc = lane & 15, ph = lane >> 4;
  const v8f vzero = {0.f, 0.f, 0.f, 0.f, 0.f, 0.f, 0.f, 0.f};
  v8f acc[4];
#pragma unroll
  for (int t = 0; t < 4; ++t) acc[t] = vzero;

  // GI over K=512, staged in two 256-wide passes (emb then context).
  for (int pass = 0; pass < 2; ++pass) {
    const float* src = pass ? context : emb;
    for (int i = tid; i < Mdim * Hdim; i += 256) {
      int m = i >> 8, k = i & 255;
      Xs[m * LDA + k] = (_Float16)src[(size_t)(bb + m) * Hdim + k];
    }
    __syncthreads();
    for (int ks = 0; ks < 8; ++ks) {
      const int k0 = ks * 32;
      v16h af = load_a_frag(Xs, LDA, m0, k0, lane);
#pragma unroll
      for (int t = 0; t < 4; ++t) {
        const int n0 = nbase + (nh * 4 + t) * 16;
        v16h bf = load_bT_frag(W_ih, 512, pass * 256 + k0, n0, lane, Gdim - 1);
        acc[t] = wmma_16x16x32(af, bf, acc[t]);
      }
    }
    __syncthreads();
  }
#pragma unroll
  for (int t = 0; t < 4; ++t) {
    const int n = nbase + (nh * 4 + t) * 16 + cc;
#pragma unroll
    for (int j = 0; j < 8; ++j)
      gi[(size_t)(bb + m0 + j + 8 * ph) * Gdim + n] = acc[t][j];
  }

  // GH over K=256 (h0).
#pragma unroll
  for (int t = 0; t < 4; ++t) acc[t] = vzero;
  for (int i = tid; i < Mdim * Hdim; i += 256) {
    int m = i >> 8, k = i & 255;
    Xs[m * LDA + k] = (_Float16)hidden[(size_t)(bb + m) * Hdim + k];
  }
  __syncthreads();
  for (int ks = 0; ks < 8; ++ks) {
    const int k0 = ks * 32;
    v16h af = load_a_frag(Xs, LDA, m0, k0, lane);
#pragma unroll
    for (int t = 0; t < 4; ++t) {
      const int n0 = nbase + (nh * 4 + t) * 16;
      v16h bf = load_bT_frag(W_hh, 256, k0, n0, lane, Gdim - 1);
      acc[t] = wmma_16x16x32(af, bf, acc[t]);
    }
  }
#pragma unroll
  for (int t = 0; t < 4; ++t) {
    const int n = nbase + (nh * 4 + t) * 16 + cc;
#pragma unroll
    for (int j = 0; j < 8; ++j)
      gh[(size_t)(bb + m0 + j + 8 * ph) * Gdim + n] = acc[t][j];
  }
}

// ---------------------------------------------------------------------------
// Kernel 3: GRU gates (elementwise) + emit hidden_new / gru_output / emb copy.
// ---------------------------------------------------------------------------
__global__ __launch_bounds__(256) void gate_kernel(
    const float* __restrict__ gi, const float* __restrict__ gh,
    const float* __restrict__ b_ih, const float* __restrict__ b_hh,
    const float* __restrict__ hidden, const float* __restrict__ emb,
    float* __restrict__ out_hidden, float* __restrict__ out_emb,
    float* __restrict__ out_gru)
{
  const int b = blockIdx.x, h = threadIdx.x;
  const size_t g = (size_t)b * Gdim;
  const float ir = gi[g + h]          + b_ih[h];
  const float iz = gi[g + Hdim + h]   + b_ih[Hdim + h];
  const float in = gi[g + 2*Hdim + h] + b_ih[2*Hdim + h];
  const float hr = gh[g + h]          + b_hh[h];
  const float hz = gh[g + Hdim + h]   + b_hh[Hdim + h];
  const float hn = gh[g + 2*Hdim + h] + b_hh[2*Hdim + h];
  const float r  = 1.f / (1.f + expf(-(ir + hr)));
  const float z  = 1.f / (1.f + expf(-(iz + hz)));
  const float nc = tanhf(in + r * hn);
  const float h0 = hidden[(size_t)b * Hdim + h];
  const float hv = (1.f - z) * nc + z * h0;
  out_hidden[(size_t)b * Hdim + h] = hv;
  out_gru[(size_t)b * Hdim + h]    = hv;
  out_emb[(size_t)b * Hdim + h]    = emb[(size_t)b * Hdim + h];
}

// ---------------------------------------------------------------------------
// Kernel 4: output = h_new @ W_out^T + b_out   (2048x256 @ 256x50000)
// Grid: (B/64, ceil(3125/8)=391). Block tile 64x128; wave: 1 M-tile x 4 N-tiles.
// ---------------------------------------------------------------------------
__global__ __launch_bounds__(256) void out_gemm_kernel(
    const float* __restrict__ hnew, const float* __restrict__ W_out,
    const float* __restrict__ b_out, float* __restrict__ out)
{
  __shared__ __attribute__((aligned(16))) _Float16 Hs[Mdim * LDA];
  const int mb = blockIdx.x * Mdim;
  const int nbase = blockIdx.y * 128;
  const int tid = threadIdx.x, lane = tid & 31, w = tid >> 5;
  const int mt = w & 3, nh = w >> 2, m0 = mt * 16;
  const int cc = lane & 15, ph = lane >> 4;

  for (int i = tid; i < Mdim * Hdim; i += 256) {
    int m = i >> 8, k = i & 255;
    Hs[m * LDA + k] = (_Float16)hnew[(size_t)(mb + m) * Hdim + k];
  }
  __syncthreads();

  const v8f vzero = {0.f, 0.f, 0.f, 0.f, 0.f, 0.f, 0.f, 0.f};
  v8f acc[4];
#pragma unroll
  for (int t = 0; t < 4; ++t) acc[t] = vzero;

  for (int ks = 0; ks < 8; ++ks) {
    const int k0 = ks * 32;
    v16h af = load_a_frag(Hs, LDA, m0, k0, lane);
#pragma unroll
    for (int t = 0; t < 4; ++t) {
      const int n0 = nbase + (nh * 4 + t) * 16;
      v16h bf = load_bT_frag(W_out, Hdim, k0, n0, lane, Vdim - 1);
      acc[t] = wmma_16x16x32(af, bf, acc[t]);
    }
  }
#pragma unroll
  for (int t = 0; t < 4; ++t) {
    const int n = nbase + (nh * 4 + t) * 16 + cc;
    if (n < Vdim) {
      const float bv = b_out[n];
#pragma unroll
      for (int j = 0; j < 8; ++j)
        out[(size_t)(mb + m0 + j + 8 * ph) * Vdim + n] = acc[t][j] + bv;
    }
  }
}

// ---------------------------------------------------------------------------
// Host launcher
// ---------------------------------------------------------------------------
extern "C" void kernel_launch(void* const* d_in, const int* in_sizes, int n_in,
                              void* d_out, int out_size, void* d_ws, size_t ws_size,
                              hipStream_t stream)
{
  const float* emb      = (const float*)d_in[0];   // (B,1,E)
  const float* hidden   = (const float*)d_in[1];   // (1,B,H)
  const float* io       = (const float*)d_in[2];   // (B,M,H)
  const float* W_inter  = (const float*)d_in[4];
  const float* b_inter  = (const float*)d_in[5];
  const float* W_hidden = (const float*)d_in[6];
  const float* b_hidden = (const float*)d_in[7];
  const float* W_scale  = (const float*)d_in[8];
  const float* b_scale  = (const float*)d_in[9];
  const float* W_lt1    = (const float*)d_in[10];
  const float* b_lt1    = (const float*)d_in[11];
  const float* W_ih     = (const float*)d_in[12];
  const float* W_hh     = (const float*)d_in[13];
  const float* b_ih     = (const float*)d_in[14];
  const float* b_hh     = (const float*)d_in[15];
  const float* W_out    = (const float*)d_in[16];
  const float* b_out    = (const float*)d_in[17];
  const int*   user_list= (const int*)d_in[19];

  float* out = (float*)d_out;
  const size_t B = Bdim, H = Hdim, V = Vdim, M = Mdim;
  float* out_output = out;                    // B*V
  float* out_hidden = out_output + B * V;     // B*H  (hidden_new)
  float* out_emb    = out_hidden + B * H;     // B*H  (emb passthrough)
  float* out_gru    = out_emb    + B * H;     // B*H  (gru_output)
  float* out_attn   = out_gru    + B * H;     // B*M  (attn_weights)

  char* ws = (char*)d_ws;
  float* ctx = (float*)ws;                               // B*H  f32
  float* gi  = (float*)(ws + B * H * 4);                 // B*768 f32
  float* gh  = (float*)(ws + B * H * 4 + B * Gdim * 4);  // B*768 f32

  attn_kernel<<<dim3(Bdim), dim3(256), 0, stream>>>(
      io, hidden, W_inter, b_inter, W_hidden, b_hidden,
      W_scale, b_scale, W_lt1, b_lt1, user_list, out_attn, ctx);

  gru_gemm_kernel<<<dim3(Bdim / 64, Gdim / 128), dim3(256), 0, stream>>>(
      emb, ctx, hidden, W_ih, W_hh, gi, gh);

  gate_kernel<<<dim3(Bdim), dim3(256), 0, stream>>>(
      gi, gh, b_ih, b_hh, hidden, emb, out_hidden, out_emb, out_gru);

  out_gemm_kernel<<<dim3(Bdim / 64, 391), dim3(256), 0, stream>>>(
      out_gru, W_out, b_out, out_output);
}
